// MultiHeadAttention_15015205667530
// MI455X (gfx1250) — matrix operand order repaired and checked by reading, not compiled, not run on hardware
//
#include <hip/hip_runtime.h>

// ---------------------------------------------------------------------------
// MI455X (gfx1250) multi-head attention, bf16 WMMA + async global->LDS staging.
//   x:[2,2048,1024] f32, w_qkv:[1024,3072] f32, w_out:[1024,1024] f32 -> out f32
// Pipeline:
//   1) f32->bf16 convert x ; transpose-convert w_qkv^T, w_out^T  (one pass)
//   2) bf16 WMMA GEMM  qkv = x @ w_qkv     (A,B^T staged via async-to-LDS,
//                                           double buffered; bf16 out)
//   3) bf16 transpose of V part of qkv -> V^T[bh][d][s]
//   4) flash attention per (b,h): K and V^T chunks async-to-LDS double
//      buffered; DPP16 row_xmask softmax reductions; bf16 out
//   5) bf16 WMMA GEMM  out = attn @ w_out  (f32 out)
// ---------------------------------------------------------------------------

typedef __bf16 bf16_t;
typedef __attribute__((ext_vector_type(16))) __bf16 v16bf;
typedef __attribute__((ext_vector_type(8)))  __bf16 v8bf;
typedef __attribute__((ext_vector_type(8)))  float  v8f;

__device__ __forceinline__ v8f wmma_bf16(v16bf a, v16bf b, v8f c) {
  return __builtin_amdgcn_wmma_f32_16x16x32_bf16(
      false, a, false, b, (short)0, c, false, false);
}

__device__ __forceinline__ v16bf ld16(const bf16_t* p) {
  union { v16bf v; v8bf h[2]; } u;
  u.h[0] = *(const v8bf*)(p);
  u.h[1] = *(const v8bf*)(p + 8);
  return u.v;
}
__device__ __forceinline__ v16bf ld8x2(const bf16_t* p0, const bf16_t* p1) {
  union { v16bf v; v8bf h[2]; } u;
  u.h[0] = *(const v8bf*)(p0);
  u.h[1] = *(const v8bf*)(p1);
  return u.v;
}

// DPP16 lane-xor within a 16-lane row (VALU pipe)
template <int XMASK>
__device__ __forceinline__ float dpp_xor(float x) {
  return __builtin_bit_cast(float,
      __builtin_amdgcn_update_dpp(0, __builtin_bit_cast(int, x),
                                  0x160 | XMASK, 0xf, 0xf, true));
}
__device__ __forceinline__ float rowmax16(float x) {
  x = fmaxf(x, dpp_xor<1>(x)); x = fmaxf(x, dpp_xor<2>(x));
  x = fmaxf(x, dpp_xor<4>(x)); x = fmaxf(x, dpp_xor<8>(x));
  return x;
}
__device__ __forceinline__ float rowsum16(float x) {
  x += dpp_xor<1>(x); x += dpp_xor<2>(x);
  x += dpp_xor<4>(x); x += dpp_xor<8>(x);
  return x;
}

// ---- CDNA5 async global->LDS copy (16B per lane), ASYNCcnt tracked --------
__device__ __forceinline__ void async_cp16(void* lds, const void* g) {
  unsigned       l = (unsigned)reinterpret_cast<unsigned long long>(lds);
  unsigned long long a = reinterpret_cast<unsigned long long>(g);
  asm volatile("global_load_async_to_lds_b128 %0, %1, off"
               :: "v"(l), "v"(a) : "memory");
}
__device__ __forceinline__ void wait_async0() {
  asm volatile("s_wait_asynccnt 0" ::: "memory");
}

// ---------------------------------------------------------------------------
// f32 -> bf16 elementwise (hardware convert), 4 elements/thread
// ---------------------------------------------------------------------------
__global__ __launch_bounds__(256) void cvt_f32_bf16(
    const float* __restrict__ in, bf16_t* __restrict__ out, int n4) {
  const int i = blockIdx.x * 256 + threadIdx.x;
  if (i < n4) {
    float4 f = ((const float4*)in)[i];
    union { bf16_t e[4]; uint2 v; } u;
    u.e[0] = (bf16_t)f.x; u.e[1] = (bf16_t)f.y;
    u.e[2] = (bf16_t)f.z; u.e[3] = (bf16_t)f.w;
    ((uint2*)out)[i] = u.v;
  }
}

// ---------------------------------------------------------------------------
// Transpose-convert: W[K][N] f32  ->  Wt[N][K] bf16  (32x32 LDS tiles)
// block (32,8), grid (N/32, K/32)
// ---------------------------------------------------------------------------
__global__ __launch_bounds__(256) void cvtT_f32_bf16(
    const float* __restrict__ in, bf16_t* __restrict__ out, int K, int N) {
  __shared__ bf16_t tile[32][33];
  const int n0 = blockIdx.x * 32, k0 = blockIdx.y * 32;
  const int tx = threadIdx.x, ty = threadIdx.y;
#pragma unroll
  for (int i = 0; i < 4; ++i)
    tile[ty + 8 * i][tx] = (bf16_t)in[(size_t)(k0 + ty + 8 * i) * N + n0 + tx];
  __syncthreads();
#pragma unroll
  for (int i = 0; i < 4; ++i)
    out[(size_t)(n0 + ty + 8 * i) * K + k0 + tx] = tile[tx][ty + 8 * i];
}

// ---------------------------------------------------------------------------
// Transpose V out of qkv:  qkv[b][s][2048 + hc] bf16 -> vt[(b*1024+hc)][s]
// block (32,8), grid (1024/32, 4096/32)
// ---------------------------------------------------------------------------
__global__ __launch_bounds__(256) void transpose_v_bf16(
    const bf16_t* __restrict__ qkv, bf16_t* __restrict__ vt) {
  __shared__ bf16_t tile[32][33];
  const int hc0 = blockIdx.x * 32, s0 = blockIdx.y * 32;
  const int b = s0 >> 11, sl0 = s0 & 2047;
  const int tx = threadIdx.x, ty = threadIdx.y;
#pragma unroll
  for (int i = 0; i < 4; ++i)
    tile[ty + 8 * i][tx] =
        qkv[(size_t)(s0 + ty + 8 * i) * 3072 + 2048 + hc0 + tx];
  __syncthreads();
#pragma unroll
  for (int i = 0; i < 4; ++i)
    vt[(size_t)(b * 1024 + hc0 + ty + 8 * i) * 2048 + sl0 + tx] =
        tile[tx][ty + 8 * i];
}

// ---------------------------------------------------------------------------
// GEMM: C[M,N] = A[M,K] * Bt[N,K]^T, bf16, f32 accumulate.
// Block 128 thr (4 waves), tile 128(M)x64(N), K-step 32, double-buffered
// async-to-LDS staging, one barrier per k-step.
// ---------------------------------------------------------------------------
#define SA 40   // padded LDS row stride (bf16): 32 data + 8 pad = 80B
#define SB 40

template <bool OUT_BF16>
__global__ __launch_bounds__(128) void gemm_wmma_bf16(
    const bf16_t* __restrict__ A, const bf16_t* __restrict__ Bt,
    void* __restrict__ Cout, int M, int N, int K) {
  __shared__ __align__(16) bf16_t lA [2][128 * SA];  // [m][k]
  __shared__ __align__(16) bf16_t lBt[2][64 * SB];   // [n][k]

  const int t    = threadIdx.x;
  const int lane = t & 31;
  const int wv   = t >> 5;
  const int hl   = lane >> 4;
  const int l16  = lane & 15;
  const int m0   = blockIdx.y * 128;
  const int n0   = blockIdx.x * 64;

  // per-thread async staging: A row t (4x16B), Bt row t>>1 (2x16B)
  auto issue = [&](int kb, int buf) {
    const bf16_t* ga = A + (size_t)(m0 + t) * K + kb;
#pragma unroll
    for (int s = 0; s < 4; ++s)
      async_cp16(&lA[buf][t * SA + s * 8], ga + s * 8);
    const int bn = t >> 1, sp = (t & 1) * 2;
    const bf16_t* gb = Bt + (size_t)(n0 + bn) * K + kb;
#pragma unroll
    for (int s = 0; s < 2; ++s)
      async_cp16(&lBt[buf][bn * SB + (sp + s) * 8], gb + (sp + s) * 8);
  };

  v8f acc0[4] = {}, acc1[4] = {};
  issue(0, 0);

  const int nsteps = K / 32;
  for (int it = 0; it < nsteps; ++it) {
    const int buf = it & 1;
    wait_async0();
    __syncthreads();                 // staged data visible; prev compute done
    if (it + 1 < nsteps) issue((it + 1) * 32, buf ^ 1);

    const bf16_t* cA = lA[buf];
    const bf16_t* cB = lBt[buf];
    const int ml = wv * 32 + l16;
    v16bf av0 = ld8x2(&cA[ml * SA + hl * 8],        &cA[ml * SA + 16 + hl * 8]);
    v16bf av1 = ld8x2(&cA[(ml + 16) * SA + hl * 8], &cA[(ml + 16) * SA + 16 + hl * 8]);
#pragma unroll
    for (int j = 0; j < 4; ++j) {
      v16bf bv = ld8x2(&cB[(j * 16 + l16) * SB + hl * 8], &cB[(j * 16 + l16) * SB + 16 + hl * 8]);
      acc0[j] = wmma_bf16(av0, bv, acc0[j]);
      acc1[j] = wmma_bf16(av1, bv, acc1[j]);
    }
  }

#pragma unroll
  for (int j = 0; j < 4; ++j) {
#pragma unroll
    for (int r = 0; r < 8; ++r) {  // C layout: m = r + 8*hl, n = l16
      const int m = m0 + wv * 32 + r + hl * 8;
      const int n = n0 + j * 16 + l16;
      if (OUT_BF16) {
        ((bf16_t*)Cout)[(size_t)m * N + n]        = (bf16_t)acc0[j][r];
        ((bf16_t*)Cout)[(size_t)(m + 16) * N + n] = (bf16_t)acc1[j][r];
      } else {
        ((float*)Cout)[(size_t)m * N + n]        = acc0[j][r];
        ((float*)Cout)[(size_t)(m + 16) * N + n] = acc1[j][r];
      }
    }
  }
}

// ---------------------------------------------------------------------------
// Flash attention. Block = one (b,h) + 64 query rows (wave -> 16 rows).
// 32-key chunks; K[key][c] and Vt[c][key] chunks async-to-LDS, double
// buffered. qkv:[B,S,3072] bf16 (q at +h*64, k at +1024+h*64); vt:[bh*64+c][s].
// ---------------------------------------------------------------------------
#define SK 72   // lK row stride: 64 data + 8 pad = 144B
#define SV 40
#define SP 40

__global__ __launch_bounds__(128) void mha_flash_attn(
    const bf16_t* __restrict__ qkv, const bf16_t* __restrict__ vt,
    bf16_t* __restrict__ attn) {
  constexpr int S = 2048, D3 = 3072, Dm = 1024, HD = 64;

  __shared__ __align__(16) bf16_t lK [2][32 * SK];
  __shared__ __align__(16) bf16_t lVt[2][64 * SV];
  __shared__ __align__(16) bf16_t lP [4][16 * SP];

  const int t    = threadIdx.x;
  const int lane = t & 31;
  const int wv   = t >> 5;
  const int hl   = lane >> 4;
  const int l16  = lane & 15;

  const int bh = blockIdx.y;
  const int b  = bh >> 4;
  const int h  = bh & 15;
  const int q0 = blockIdx.x * 64 + wv * 16;

  const bf16_t* qbase = qkv + (size_t)b * S * D3 + h * HD;
  const bf16_t* kbase = qbase + Dm;
  const bf16_t* vtb   = vt + (size_t)bh * HD * S;   // [c][s]

  // staging: K chunk rows 128B (8x16B) -> thread t: key=t>>2, 2 segs;
  //          Vt chunk rows 64B (4x16B) -> thread t: c=t>>1, 2 segs.
  auto issue = [&](int kc, int buf) {
    const int key = t >> 2, ks = (t & 3) * 2;
    const bf16_t* gk = kbase + (size_t)(kc + key) * D3;
#pragma unroll
    for (int s = 0; s < 2; ++s)
      async_cp16(&lK[buf][key * SK + (ks + s) * 8], gk + (ks + s) * 8);
    const int c = t >> 1, vs = (t & 1) * 2;
    const bf16_t* gv = vtb + (size_t)c * S + kc;
#pragma unroll
    for (int s = 0; s < 2; ++s)
      async_cp16(&lVt[buf][c * SV + (vs + s) * 8], gv + (vs + s) * 8);
  };

  // Q operands (A layout), d=64 -> two 32-deep K-steps; loaded once
  v16bf qa[2];
  {
    const bf16_t* qrow = qbase + (size_t)(q0 + l16) * D3;
#pragma unroll
    for (int ks = 0; ks < 2; ++ks)
      qa[ks] = ld8x2(qrow + ks * 32 + hl * 8, qrow + ks * 32 + 16 + hl * 8);
  }

  const float scl = 0.125f * 1.44269504088896340736f;  // 1/sqrt(64)*log2(e)
  float mstate[8], lstate[8];
  v8f o[4] = {};
#pragma unroll
  for (int r = 0; r < 8; ++r) { mstate[r] = -1e30f; lstate[r] = 0.f; }

  issue(0, 0);
  for (int it = 0; it < S / 32; ++it) {
    const int buf = it & 1;
    wait_async0();
    __syncthreads();
    if (it + 1 < S / 32) issue((it + 1) * 32, buf ^ 1);

    const bf16_t* cK = lK[buf];
    const bf16_t* cV = lVt[buf];

    // scores: two 16x16 key tiles, each over two 32-deep K-steps
    v8f s0v = {}, s1v = {};
#pragma unroll
    for (int ks = 0; ks < 2; ++ks) {
      v16bf b0 = ld8x2(&cK[(l16)      * SK + ks * 32 + hl * 8], &cK[(l16)      * SK + ks * 32 + 16 + hl * 8]);
      v16bf b1 = ld8x2(&cK[(16 + l16) * SK + ks * 32 + hl * 8], &cK[(16 + l16) * SK + ks * 32 + 16 + hl * 8]);
      s0v = wmma_bf16(qa[ks], b0, s0v);
      s1v = wmma_bf16(qa[ks], b1, s1v);
    }

    // online softmax; DPP16 row reductions on the VALU pipe
    float p0[8], p1[8];
#pragma unroll
    for (int r = 0; r < 8; ++r) {
      float t0 = s0v[r] * scl, t1 = s1v[r] * scl;
      float rm = rowmax16(fmaxf(t0, t1));
      float mnew  = fmaxf(mstate[r], rm);
      float alpha = exp2f(mstate[r] - mnew);
      p0[r] = exp2f(t0 - mnew);
      p1[r] = exp2f(t1 - mnew);
      float rs = rowsum16(p0[r] + p1[r]);
      lstate[r] = lstate[r] * alpha + rs;
      mstate[r] = mnew;
#pragma unroll
      for (int g = 0; g < 4; ++g) o[g][r] *= alpha;
    }

    // C-layout -> A-layout for P via per-wave LDS tile
    bf16_t* P = lP[wv];
#pragma unroll
    for (int r = 0; r < 8; ++r) {
      const int m = r + hl * 8;
      P[m * SP + l16]      = (bf16_t)p0[r];
      P[m * SP + 16 + l16] = (bf16_t)p1[r];
    }
    v16bf pa = ld8x2(&P[l16 * SP + hl * 8], &P[l16 * SP + 16 + hl * 8]);

    // O += P * V  (4 channel groups of 16)
#pragma unroll
    for (int g = 0; g < 4; ++g) {
      v16bf vb = ld8x2(&cV[(g * 16 + l16) * SV + hl * 8], &cV[(g * 16 + l16) * SV + 16 + hl * 8]);
      o[g] = wmma_bf16(pa, vb, o[g]);
    }
  }

  bf16_t* obase = attn + (size_t)b * S * Dm + h * HD;
#pragma unroll
  for (int r = 0; r < 8; ++r) {
    const float inv = 1.0f / lstate[r];
    bf16_t* orow = obase + (size_t)(q0 + r + hl * 8) * Dm;
#pragma unroll
    for (int g = 0; g < 4; ++g) orow[g * 16 + l16] = (bf16_t)(o[g][r] * inv);
  }
}

// ---------------------------------------------------------------------------
extern "C" void kernel_launch(void* const* d_in, const int* in_sizes, int n_in,
                              void* d_out, int out_size, void* d_ws, size_t ws_size,
                              hipStream_t stream) {
  (void)in_sizes; (void)n_in; (void)out_size; (void)ws_size;
  const float* x     = (const float*)d_in[0];  // 4096 x 1024
  const float* w_qkv = (const float*)d_in[1];  // [1024,3072]
  const float* w_out = (const float*)d_in[2];  // [1024,1024]
  float* out = (float*)d_out;                  // 4096 x 1024 f32

  const int M = 4096, K = 1024, N3 = 3072, N1 = 1024;

  // workspace (bf16): x | w_qkv^T | w_out^T | qkv | v^T | attn  (~56 MB)
  bf16_t* xb    = (bf16_t*)d_ws;
  bf16_t* wqkvT = xb    + (size_t)M * K;
  bf16_t* woutT = wqkvT + (size_t)N3 * K;
  bf16_t* qkvb  = woutT + (size_t)N1 * K;
  bf16_t* vtb   = qkvb  + (size_t)M * N3;     // [2*16*64][2048]
  bf16_t* attnb = vtb   + (size_t)2 * 1024 * 2048;

  const int nx = M * K / 4;
  cvt_f32_bf16 <<<(nx + 255) / 256, 256, 0, stream>>>(x, xb, nx);
  cvtT_f32_bf16<<<dim3(N3 / 32, K / 32), dim3(32, 8), 0, stream>>>(w_qkv, wqkvT, K, N3);
  cvtT_f32_bf16<<<dim3(N1 / 32, K / 32), dim3(32, 8), 0, stream>>>(w_out, woutT, K, N1);

  dim3 blk(128);
  gemm_wmma_bf16<true ><<<dim3(N3 / 64, M / 128), blk, 0, stream>>>(xb, wqkvT, qkvb, M, N3, K);
  transpose_v_bf16<<<dim3(1024 / 32, 4096 / 32), dim3(32, 8), 0, stream>>>(qkvb, vtb);
  mha_flash_attn<<<dim3(2048 / 64, 32), blk, 0, stream>>>(qkvb, vtb, attnb);
  gemm_wmma_bf16<false><<<dim3(N1 / 64, M / 128), blk, 0, stream>>>(attnb, woutT, out, M, N1, K);
}
